// NewDisentangler_6_42975442764089
// MI455X (gfx1250) — compile-verified, matrix operand order checked by reading, not run on hardware
//
#include <hip/hip_runtime.h>
#include <hip/hip_bf16.h>
#include <math.h>

typedef __attribute__((ext_vector_type(16))) __bf16 v16bf;
typedef __attribute__((ext_vector_type(4)))  __bf16 v4bf;
typedef __attribute__((ext_vector_type(8)))  float  v8f;

#define NB   16
#define CDIM 1024
#define HWSZ 1024
#define CQKD 128

__device__ __forceinline__ void atomicMaxF(float* addr, float v) {
    // sign-aware int trick: works for mixed-sign values, init = -FLT_MAX
    if (v >= 0.f) atomicMax((int*)addr, __float_as_int(v));
    else          atomicMin((unsigned int*)addr, (unsigned int)__float_as_int(v));
}

// ---------------------------------------------------------------------------
// init: zero conv accumulator, set global_feats to -inf
// ---------------------------------------------------------------------------
__global__ __launch_bounds__(256) void init_k(float* __restrict__ aBuf,
                                              float* __restrict__ gf) {
    int i = blockIdx.x * 256 + threadIdx.x;   // 16384 threads
    aBuf[i] = 0.f;
    gf[i]   = -3.4e38f;
}

// ---------------------------------------------------------------------------
// 3x3 conv head, C summed, partial over c-chunks via atomicAdd
// grid (16, 8), block 256
// ---------------------------------------------------------------------------
__global__ __launch_bounds__(256) void conv_k(const float* __restrict__ x,
                                              const float* __restrict__ hw,
                                              float* __restrict__ aBuf) {
    __shared__ float plane[1024];
    const int n = blockIdx.x, cb = blockIdx.y, tid = threadIdx.x;
    float acc[4] = {0.f, 0.f, 0.f, 0.f};
    for (int cc = 0; cc < 128; ++cc) {
        const int c = cb * 128 + cc;
        __syncthreads();
        #pragma unroll
        for (int j = 0; j < 4; ++j)
            plane[tid + j * 256] = x[((long)(n * CDIM + c)) * HWSZ + tid + j * 256];
        __syncthreads();
        float wv[9];
        #pragma unroll
        for (int t = 0; t < 9; ++t) wv[t] = hw[c * 9 + t];
        #pragma unroll
        for (int j = 0; j < 4; ++j) {
            const int p = tid + j * 256;
            const int py = p >> 5, px = p & 31;
            float s = 0.f;
            #pragma unroll
            for (int dy = -1; dy <= 1; ++dy) {
                const int yy = py + dy;
                if (yy < 0 || yy > 31) continue;
                #pragma unroll
                for (int dx = -1; dx <= 1; ++dx) {
                    const int xx = px + dx;
                    if (xx < 0 || xx > 31) continue;
                    s += wv[(dy + 1) * 3 + (dx + 1)] * plane[yy * 32 + xx];
                }
            }
            acc[j] += s;
        }
    }
    #pragma unroll
    for (int j = 0; j < 4; ++j)
        atomicAdd(&aBuf[n * 1024 + tid + j * 256], acc[j]);
}

// ---------------------------------------------------------------------------
// batch-norm statistics over all 16384 conv outputs; 1 block of 256
// ---------------------------------------------------------------------------
__global__ __launch_bounds__(256) void stats_k(const float* __restrict__ aBuf,
                                               float* __restrict__ stats) {
    __shared__ float s1[256], s2[256];
    const int tid = threadIdx.x;
    float a1 = 0.f, a2 = 0.f;
    for (int i = tid; i < 16384; i += 256) { float v = aBuf[i]; a1 += v; a2 += v * v; }
    s1[tid] = a1; s2[tid] = a2; __syncthreads();
    for (int s = 128; s > 0; s >>= 1) {
        if (tid < s) { s1[tid] += s1[tid + s]; s2[tid] += s2[tid + s]; }
        __syncthreads();
    }
    if (tid == 0) {
        const float mu  = s1[0] * (1.f / 16384.f);
        const float var = s2[0] * (1.f / 16384.f) - mu * mu;
        stats[0] = mu;
        stats[1] = rsqrtf(var + 1e-5f);
    }
}

// ---------------------------------------------------------------------------
// ccam = sigmoid(BN(a));  writes both scratch copy and output copy
// ---------------------------------------------------------------------------
__global__ __launch_bounds__(256) void ccam_k(const float* __restrict__ aBuf,
                                              const float* __restrict__ stats,
                                              const float* __restrict__ bnw,
                                              const float* __restrict__ bnb,
                                              float* __restrict__ cm,
                                              float* __restrict__ outC) {
    const int i = blockIdx.x * 256 + threadIdx.x;   // 16384
    const float v = (aBuf[i] - stats[0]) * stats[1] * bnw[0] + bnb[0];
    const float s = 1.f / (1.f + __expf(-v));
    cm[i] = s;
    outC[i] = s;
}

// ---------------------------------------------------------------------------
// fg[n,c] = (cm . x[n,c,:]) / hw ;  bg = mean_h(x) - fg
// grid (16, 128), block 256: wave per channel, lanes stride h (coalesced)
// ---------------------------------------------------------------------------
__global__ __launch_bounds__(256) void fgbg_k(const float* __restrict__ x,
                                              const float* __restrict__ cm,
                                              float* __restrict__ fg,
                                              float* __restrict__ bg) {
    __shared__ float cs[1024];
    const int n = blockIdx.x, tid = threadIdx.x;
    #pragma unroll
    for (int j = 0; j < 4; ++j) cs[tid + j * 256] = cm[n * 1024 + tid + j * 256];
    __syncthreads();
    const int wv = tid >> 5, lane = tid & 31;
    const int c = blockIdx.y * 8 + wv;
    const float* xr = x + ((long)n * CDIM + c) * HWSZ;
    float fa = 0.f, xs = 0.f;
    #pragma unroll 4
    for (int i = 0; i < 32; ++i) {
        const int h = i * 32 + lane;
        const float xv = xr[h];
        fa += cs[h] * xv; xs += xv;
    }
    #pragma unroll
    for (int s = 16; s; s >>= 1) {
        fa += __shfl_xor(fa, s, 32);
        xs += __shfl_xor(xs, s, 32);
    }
    if (lane == 0) {
        const float fgv = fa * (1.f / 1024.f);
        fg[n * 1024 + c] = fgv;
        bg[n * 1024 + c] = xs * (1.f / 1024.f) - fgv;
    }
}

// ---------------------------------------------------------------------------
// row softmax on bf16 S, in place. grid = 16*1024 blocks of 256.
// contiguous 4 elems per thread -> single b64 load/store
// ---------------------------------------------------------------------------
__global__ __launch_bounds__(256) void softmax_k(__bf16* __restrict__ S) {
    __shared__ float red[256];
    const long row = blockIdx.x;
    __bf16* p = S + row * 1024;
    const int tid = threadIdx.x;
    float v[4];
    #pragma unroll
    for (int j = 0; j < 4; ++j) v[j] = (float)p[tid * 4 + j];
    float mx = fmaxf(fmaxf(v[0], v[1]), fmaxf(v[2], v[3]));
    red[tid] = mx; __syncthreads();
    for (int s = 128; s > 0; s >>= 1) {
        if (tid < s) red[tid] = fmaxf(red[tid], red[tid + s]);
        __syncthreads();
    }
    mx = red[0]; __syncthreads();
    float sm = 0.f;
    #pragma unroll
    for (int j = 0; j < 4; ++j) { v[j] = __expf(v[j] - mx); sm += v[j]; }
    red[tid] = sm; __syncthreads();
    for (int s = 128; s > 0; s >>= 1) {
        if (tid < s) red[tid] += red[tid + s];
        __syncthreads();
    }
    const float inv = 1.f / red[0];
    v4bf pk;
    #pragma unroll
    for (int j = 0; j < 4; ++j) pk[j] = (__bf16)(v[j] * inv);
    *reinterpret_cast<v4bf*>(&p[tid * 4]) = pk;
}

// ---------------------------------------------------------------------------
// Batched bf16 WMMA GEMM, 128x128 workgroup tile, K-step 32.
// Double-buffered LDS + register staging; one barrier per K-step.
// A staged 1x16 per thread (b128 LDS stores); B staged as 4x4 micro-tile
// (contiguous-N global loads, contiguous-K v4bf = ds_store_b64 commits).
// 8 wave32 waves: wave (wm,wn) owns a 32x64 subtile = 2x4 accumulators.
// MODE 0: out bf16 [M,N] (+bias)   MODE 1: out bf16 [N,M] (+bias)
// MODE 2: epilogue  max_i(gamma*acc + x[c,i])  -> atomicMax into gf
// ---------------------------------------------------------------------------
template<typename AT, typename BT, int MODE, bool HAS_BIAS>
__global__ __launch_bounds__(256) void gemm_wmma(
    const AT* __restrict__ A, const BT* __restrict__ B,
    const float* __restrict__ bias, __bf16* __restrict__ out,
    long aStride, long bStride, long oStride,
    int M, int N, int K,
    const float* __restrict__ gamma, const float* __restrict__ xin,
    float* __restrict__ gf)
{
    __shared__ __bf16 ldsA[2][128 * 40];   // [m][k], padded row stride 40
    __shared__ __bf16 ldsB[2][128 * 40];   // [n][k] (B stored transposed)
    const int tid  = threadIdx.x;
    const int lane = tid & 31;
    const int wv   = tid >> 5;
    const int wm   = wv & 3, wn = wv >> 2;
    const int n    = blockIdx.z;
    const int bM0  = blockIdx.x * 128;
    const int bN0  = blockIdx.y * 128;
    const AT* Ab = A + (long)n * aStride;
    const BT* Bb = B + (long)n * bStride;

    v8f acc[2][4];
    #pragma unroll
    for (int i = 0; i < 2; ++i)
        #pragma unroll
        for (int j = 0; j < 4; ++j) acc[i][j] = (v8f){0.f,0.f,0.f,0.f,0.f,0.f,0.f,0.f};

    const int la_row = tid >> 1, la_k = (tid & 1) * 16;   // A: 1x16 strip
    const int kg = tid >> 5, ng = tid & 31;               // B: 4x4 micro-tile
    const int m = lane & 15, g = lane >> 4;
    const int nk = K >> 5;

    // ---- preload tile 0 into LDS buffer 0 ----
    {
        const AT* srcA = Ab + (long)(bM0 + la_row) * K + la_k;
        __bf16* dstA = &ldsA[0][la_row * 40 + la_k];
        #pragma unroll
        for (int j = 0; j < 16; ++j) dstA[j] = (__bf16)(float)srcA[j];
        #pragma unroll
        for (int i = 0; i < 4; ++i) {
            const BT* srcB = Bb + (long)(kg * 4 + i) * N + bN0 + ng * 4;
            #pragma unroll
            for (int j = 0; j < 4; ++j)
                ldsB[0][(ng * 4 + j) * 40 + kg * 4 + i] = (__bf16)(float)srcB[j];
        }
    }
    __syncthreads();

    for (int kt = 0; kt < nk; ++kt) {
        const int cur = kt & 1, nxt = cur ^ 1;
        const bool more = (kt + 1) < nk;     // uniform

        // ---- stage tile kt+1 from global into registers (overlaps WMMA) ----
        float ra[16], rb[4][4];
        if (more) {
            const AT* srcA = Ab + (long)(bM0 + la_row) * K + (kt + 1) * 32 + la_k;
            #pragma unroll
            for (int j = 0; j < 16; ++j) ra[j] = (float)srcA[j];
            #pragma unroll
            for (int i = 0; i < 4; ++i) {
                const BT* srcB = Bb + (long)((kt + 1) * 32 + kg * 4 + i) * N + bN0 + ng * 4;
                #pragma unroll
                for (int j = 0; j < 4; ++j) rb[i][j] = (float)srcB[j];
            }
        }

        // ---- fragments from LDS buffer cur + 8 WMMAs ----
        v16bf af[2], bfr[4];
        #pragma unroll
        for (int rt = 0; rt < 2; ++rt) {
            const __bf16* p = &ldsA[cur][(wm * 32 + rt * 16 + m) * 40];
            #pragma unroll
            for (int t = 0; t < 8; ++t) af[rt][t] = p[g * 8 + t];
            #pragma unroll
            for (int t = 0; t < 8; ++t) af[rt][8 + t] = p[16 + g * 8 + t];
        }
        #pragma unroll
        for (int ct = 0; ct < 4; ++ct) {
            const __bf16* p = &ldsB[cur][(wn * 64 + ct * 16 + m) * 40];
            #pragma unroll
            for (int t = 0; t < 8; ++t) bfr[ct][t] = p[g * 8 + t];
            #pragma unroll
            for (int t = 0; t < 8; ++t) bfr[ct][8 + t] = p[16 + g * 8 + t];
        }
        #pragma unroll
        for (int rt = 0; rt < 2; ++rt)
            #pragma unroll
            for (int ct = 0; ct < 4; ++ct)
                acc[rt][ct] = __builtin_amdgcn_wmma_f32_16x16x32_bf16(
                    false, af[rt], false, bfr[ct], (short)0, acc[rt][ct], false, false);

        // ---- commit staged tile to the other LDS buffer ----
        if (more) {
            __bf16* dstA = &ldsA[nxt][la_row * 40 + la_k];
            #pragma unroll
            for (int j = 0; j < 16; ++j) dstA[j] = (__bf16)ra[j];
            #pragma unroll
            for (int j = 0; j < 4; ++j) {
                v4bf pk;
                #pragma unroll
                for (int i = 0; i < 4; ++i) pk[i] = (__bf16)rb[i][j];
                *reinterpret_cast<v4bf*>(&ldsB[nxt][(ng * 4 + j) * 40 + kg * 4]) = pk;
            }
        }
        __syncthreads();
    }

    if (MODE == 2) {
        const float gm = gamma[0];
        const long xn = (long)n * CDIM * HWSZ;
        #pragma unroll
        for (int ct = 0; ct < 4; ++ct) {
            const int gN = bN0 + wn * 64 + ct * 16 + m;     // channel c
            float mx = -3.4e38f;
            #pragma unroll
            for (int rt = 0; rt < 2; ++rt)
                #pragma unroll
                for (int r = 0; r < 8; ++r) {
                    const int gM = bM0 + wm * 32 + rt * 16 + g * 8 + r;   // pixel i
                    const float v = gm * acc[rt][ct][r] + xin[xn + (long)gN * HWSZ + gM];
                    mx = fmaxf(mx, v);
                }
            mx = fmaxf(mx, __shfl_xor(mx, 16, 32));
            if (lane < 16) atomicMaxF(&gf[n * CDIM + gN], mx);
        }
    } else {
        // hoisted bias (no per-element branching)
        float bv[2][8];
        #pragma unroll
        for (int rt = 0; rt < 2; ++rt)
            #pragma unroll
            for (int r = 0; r < 8; ++r)
                bv[rt][r] = HAS_BIAS ? bias[bM0 + wm * 32 + rt * 16 + g * 8 + r] : 0.f;

        __bf16* ob = out + (long)n * oStride;
        #pragma unroll
        for (int rt = 0; rt < 2; ++rt)
            #pragma unroll
            for (int ct = 0; ct < 4; ++ct)
                #pragma unroll
                for (int r = 0; r < 8; ++r) {
                    const int gM = bM0 + wm * 32 + rt * 16 + g * 8 + r;
                    const int gN = bN0 + wn * 64 + ct * 16 + m;
                    const float v = acc[rt][ct][r] + bv[rt][r];
                    if (MODE == 0) ob[(long)gM * N + gN] = (__bf16)v;
                    else           ob[(long)gN * M + gM] = (__bf16)v;
                }
    }
}

// ---------------------------------------------------------------------------
extern "C" void kernel_launch(void* const* d_in, const int* in_sizes, int n_in,
                              void* d_out, int out_size, void* d_ws, size_t ws_size,
                              hipStream_t stream) {
    const float* x      = (const float*)d_in[0];
    const float* head_w = (const float*)d_in[1];
    const float* bn_w   = (const float*)d_in[2];
    const float* bn_b   = (const float*)d_in[3];
    const float* q_w    = (const float*)d_in[4];
    const float* q_b    = (const float*)d_in[5];
    const float* k_w    = (const float*)d_in[6];
    const float* k_b    = (const float*)d_in[7];
    const float* v_w    = (const float*)d_in[8];
    const float* v_b    = (const float*)d_in[9];
    const float* gamma  = (const float*)d_in[10];

    float* out   = (float*)d_out;
    float* outFg = out;                 // [16,1024]
    float* outBg = out + 16384;         // [16,1024]
    float* outCc = out + 32768;         // [16,1,32,32]
    float* outGf = out + 49152;         // [16,1024]

    char* w = (char*)d_ws;
    size_t off = 0;
    __bf16* qT = (__bf16*)(w + off); off += (size_t)NB * HWSZ * CQKD * 2;   // [n][h][o]
    __bf16* kM = (__bf16*)(w + off); off += (size_t)NB * CQKD * HWSZ * 2;   // [n][o][h]
    __bf16* vT = (__bf16*)(w + off); off += (size_t)NB * HWSZ * CDIM * 2;   // [n][h][c]
    __bf16* S  = (__bf16*)(w + off); off += (size_t)NB * HWSZ * HWSZ * 2;   // [n][i][j]
    float* aBuf  = (float*)(w + off); off += 16384 * 4;
    float* cmBuf = (float*)(w + off); off += 16384 * 4;
    float* stats = (float*)(w + off); off += 256;

    // activation head
    init_k <<<64, 256, 0, stream>>>(aBuf, outGf);
    conv_k <<<dim3(16, 8), 256, 0, stream>>>(x, head_w, aBuf);
    stats_k<<<1, 256, 0, stream>>>(aBuf, stats);
    ccam_k <<<64, 256, 0, stream>>>(aBuf, stats, bn_w, bn_b, cmBuf, outCc);
    fgbg_k <<<dim3(16, 128), 256, 0, stream>>>(x, cmBuf, outFg, outBg);

    const long XB = (long)CDIM * HWSZ;          // per-n x elems
    // q = q_w @ x + q_b  -> stored transposed [h][o]
    gemm_wmma<float, float, 1, true><<<dim3(1, 8, 16), 256, 0, stream>>>(
        q_w, x, q_b, qT, 0, XB, (long)HWSZ * CQKD, CQKD, HWSZ, CDIM,
        nullptr, nullptr, nullptr);
    // k = k_w @ x + k_b  -> stored [o][h]
    gemm_wmma<float, float, 0, true><<<dim3(1, 8, 16), 256, 0, stream>>>(
        k_w, x, k_b, kM, 0, XB, (long)CQKD * HWSZ, CQKD, HWSZ, CDIM,
        nullptr, nullptr, nullptr);
    // v = v_w @ x + v_b  -> stored transposed [h][c]
    gemm_wmma<float, float, 1, true><<<dim3(8, 8, 16), 256, 0, stream>>>(
        v_w, x, v_b, vT, 0, XB, (long)HWSZ * CDIM, CDIM, HWSZ, CDIM,
        nullptr, nullptr, nullptr);
    // S = qT @ k  (per n), [i][j]
    gemm_wmma<__bf16, __bf16, 0, false><<<dim3(8, 8, 16), 256, 0, stream>>>(
        qT, kM, nullptr, S,
        (long)HWSZ * CQKD, (long)CQKD * HWSZ, (long)HWSZ * HWSZ,
        HWSZ, HWSZ, CQKD, nullptr, nullptr, nullptr);
    // P = softmax rows of S, in place
    softmax_k<<<NB * HWSZ, 256, 0, stream>>>(S);
    // out_t = P @ vT ; epilogue gamma*acc + x, column max -> global_feats
    gemm_wmma<__bf16, __bf16, 2, false><<<dim3(8, 8, 16), 256, 0, stream>>>(
        S, vT, nullptr, nullptr,
        (long)HWSZ * HWSZ, (long)HWSZ * CDIM, 0,
        HWSZ, CDIM, HWSZ, gamma, x, outGf);
}
